// AttentionXMLModified_63376537419962
// MI455X (gfx1250) — matrix operand order repaired and checked by reading, not compiled, not run on hardware
//
#include <hip/hip_runtime.h>
#include <hip/hip_bf16.h>
#include <math.h>

typedef unsigned short u16;
typedef unsigned int   u32;
typedef __attribute__((ext_vector_type(16))) __bf16 v16bf;
typedef __attribute__((ext_vector_type(8)))  float  v8f;

#define B_   16
#define S_   256
#define D_   512
#define H_   1024
#define L_   2
#define F_   2048
#define HID_ 256
#define G4   1024   // 4*HID

__device__ __forceinline__ u16 f2bf(float x) {
  union { float f; u32 u; } t; t.f = x;
  u32 r = t.u + 0x7FFFu + ((t.u >> 16) & 1u);   // round-to-nearest-even
  return (u16)(r >> 16);
}
__device__ __forceinline__ float sigmoidf_(float x) { return 1.0f / (1.0f + __expf(-x)); }
__device__ __forceinline__ float geluf_(float x)    { return 0.5f * x * (1.0f + erff(x * 0.70710678118654752f)); }

// ---------- WMMA bf16 fragment helpers (16x16x32, wave32) ----------
// A fragment: 16(M) x 32(K), row-major halfwords, row stride ld (halves).
// Per-lane data = two contiguous 16B chunks: halves [half*8, +8) and [16+half*8, +8).
__device__ __forceinline__ v16bf load_frag_a(const u16* p, int ld) {
  int lane = threadIdx.x & 31;
  int m = lane & 15, half = lane >> 4;
  const u16* row = p + m * ld + half * 8;
  union { v16bf v; uint4 x[2]; } u;
  u.x[0] = *(const uint4*)(row);
  u.x[1] = *(const uint4*)(row + 16);
  return u.v;
}
// B fragment: 32(K) x 16(N), stored n-major (row = N, stride ld halves).
// Per-lane data = two contiguous 16B chunks: halves [half*8, +8) and [16+half*8, +8).
__device__ __forceinline__ v16bf load_frag_b(const u16* p, int ld) {
  int lane = threadIdx.x & 31;
  int n = lane & 15, half = lane >> 4;
  const u16* row = p + n * ld + half * 8;
  union { v16bf v; uint4 x[2]; } u;
  u.x[0] = *(const uint4*)(row);
  u.x[1] = *(const uint4*)(row + 16);
  return u.v;
}
__device__ __forceinline__ v8f wmma_bf16(v16bf a, v16bf b, v8f c) {
  return __builtin_amdgcn_wmma_f32_16x16x32_bf16(false, a, false, b, (short)0, c, false, false);
}

// ---------- Generic batched GEMM: C = A * op(B) (+bias)(+gelu) ----------
// A: [M,K] row-major f32. TRANS_B: B is [N,K] (compute A*B^T) else [K,N].
// M%128==0, N%128==0, K%32==0. 256 threads = 8 waves (4 M x 2 N), 128x128 tile.
// EPI: 0 none, 1 +bias[n], 2 +bias[n] then GELU.
template<bool TRANS_B, int EPI>
__global__ __launch_bounds__(256) void gemm_wmma(
    const float* __restrict__ A, const float* __restrict__ Bm,
    const float* __restrict__ bias, float* __restrict__ C,
    int M, int N, int K, long long sA, long long sB, long long sC)
{
  __shared__ u16 As[128 * 40];   // [m][k], stride 40 halves = 80B (16B-divisible)
  __shared__ u16 Bs[128 * 40];   // [n][k]
  const int z = blockIdx.z;
  A  += (long long)z * sA;
  Bm += (long long)z * sB;
  C  += (long long)z * sC;
  const int bm = blockIdx.x * 128, bn = blockIdx.y * 128;
  const int tid = threadIdx.x, wave = tid >> 5;
  const int wm = wave & 3, wn = wave >> 2;
  const v8f vz = {0.f,0.f,0.f,0.f,0.f,0.f,0.f,0.f};
  v8f acc[2][4];
#pragma unroll
  for (int i = 0; i < 2; ++i)
#pragma unroll
    for (int j = 0; j < 4; ++j) acc[i][j] = vz;

  for (int k0 = 0; k0 < K; k0 += 32) {
    { // stage A 128x32 (f32 -> bf16)
      int row = tid >> 1, cb = (tid & 1) * 16;
      const float4* src = (const float4*)(A + (long long)(bm + row) * K + k0 + cb);
      u16* dst = As + row * 40 + cb;
#pragma unroll
      for (int j = 0; j < 4; ++j) {
        float4 f = src[j];
        dst[j*4+0]=f2bf(f.x); dst[j*4+1]=f2bf(f.y); dst[j*4+2]=f2bf(f.z); dst[j*4+3]=f2bf(f.w);
      }
    }
    if (TRANS_B) { // B[N,K]: copy rows directly into [n][k]
      int n = tid >> 1, cb = (tid & 1) * 16;
      const float4* src = (const float4*)(Bm + (long long)(bn + n) * K + k0 + cb);
      u16* dst = Bs + n * 40 + cb;
#pragma unroll
      for (int j = 0; j < 4; ++j) {
        float4 f = src[j];
        dst[j*4+0]=f2bf(f.x); dst[j*4+1]=f2bf(f.y); dst[j*4+2]=f2bf(f.z); dst[j*4+3]=f2bf(f.w);
      }
    } else {       // B[K,N]: transpose into [n][k]
      int k = tid >> 3, nb = (tid & 7) * 16;
      const float4* src = (const float4*)(Bm + (long long)(k0 + k) * N + bn + nb);
#pragma unroll
      for (int j = 0; j < 4; ++j) {
        float4 f = src[j];
        Bs[(nb + j*4+0)*40 + k] = f2bf(f.x);
        Bs[(nb + j*4+1)*40 + k] = f2bf(f.y);
        Bs[(nb + j*4+2)*40 + k] = f2bf(f.z);
        Bs[(nb + j*4+3)*40 + k] = f2bf(f.w);
      }
    }
    __syncthreads();
    v16bf a0 = load_frag_a(As + (wm*32 + 0 ) * 40, 40);
    v16bf a1 = load_frag_a(As + (wm*32 + 16) * 40, 40);
#pragma unroll
    for (int j = 0; j < 4; ++j) {
      v16bf b = load_frag_b(Bs + (wn*64 + j*16) * 40, 40);
      acc[0][j] = wmma_bf16(a0, b, acc[0][j]);
      acc[1][j] = wmma_bf16(a1, b, acc[1][j]);
    }
    __syncthreads();
  }
  const int lane = tid & 31, half = lane >> 4, nlo = lane & 15;
#pragma unroll
  for (int i = 0; i < 2; ++i)
#pragma unroll
    for (int j = 0; j < 4; ++j) {
      int n = bn + wn*64 + j*16 + nlo;
      float bv = (EPI >= 1) ? bias[n] : 0.0f;
#pragma unroll
      for (int r = 0; r < 8; ++r) {
        int m = bm + wm*32 + i*16 + r + half*8;
        float v = acc[i][j][r] + bv;
        if (EPI == 2) v = geluf_(v);
        C[(long long)m * N + n] = v;
      }
    }
}

// ---------- Persistent LSTM scan: one workgroup per direction ----------
// Per step: g[16,1024] = pre[:,t,:] + h(bf16,LDS) @ Whh^T (WMMA), gates via LDS,
// c kept in registers, h written back to LDS (bf16) and features (f32).
// An opaque per-iteration zero (asm volatile) keeps the Whh fragment loads
// inside the step loop (regular cached global_load_b128, no hoist -> no spill,
// no volatile serialization).
__global__ __launch_bounds__(1024) void lstm_scan(
    const float* __restrict__ pre_f, const float* __restrict__ pre_b,
    const u16* __restrict__ whh_f, const u16* __restrict__ whh_b,
    float* __restrict__ features)
{
  extern __shared__ char smem[];
  float* gates = (float*)smem;                 // 16*1024 f32 = 64KB
  u16*   hbuf  = (u16*)(smem + 16*1024*4);     // 16*256 bf16 = 8KB
  const int dir = blockIdx.x;
  const float* pre = dir ? pre_b : pre_f;
  const u16*   whh = dir ? whh_b : whh_f;      // [1024][256] bf16
  const int tid = threadIdx.x, wave = tid >> 5, lane = tid & 31;
  const int half = lane >> 4, nlo = lane & 15;

  for (int i = tid; i < 16*256; i += 1024) hbuf[i] = 0;
  float cst[4] = {0.f, 0.f, 0.f, 0.f};
  __syncthreads();

  for (int step = 0; step < S_; ++step) {
    const int t = dir ? (S_ - 1 - step) : step;
    // Opaque zero: defeats loop-invariant hoisting of the weight loads.
    u32 opaque;
    asm volatile("s_mov_b32 %0, 0" : "=s"(opaque));
    const u16* whh_s = whh + opaque;

    v8f acc[2];
#pragma unroll
    for (int j = 0; j < 2; ++j) {
      int n = wave*32 + j*16 + nlo;
#pragma unroll
      for (int r = 0; r < 8; ++r) {
        int b = r + half*8;
        acc[j][r] = pre[((long long)(b*S_ + t))*G4 + n];
      }
    }
#pragma unroll
    for (int kc = 0; kc < 8; ++kc) {
      v16bf a = load_frag_a(hbuf + kc*32, 256);          // h: [batch][hid]
#pragma unroll
      for (int j = 0; j < 2; ++j) {
        v16bf b = load_frag_b(whh_s + (long long)(wave*32 + j*16)*256 + kc*32, 256);
        acc[j] = wmma_bf16(a, b, acc[j]);
      }
    }
#pragma unroll
    for (int j = 0; j < 2; ++j) {
      int n = wave*32 + j*16 + nlo;
#pragma unroll
      for (int r = 0; r < 8; ++r) gates[(r + half*8)*G4 + n] = acc[j][r];
    }
    __syncthreads();
#pragma unroll
    for (int i = 0; i < 4; ++i) {
      int e = tid + i*1024;
      int b = e >> 8, n = e & 255;
      float gi = gates[b*G4 + n];
      float gf = gates[b*G4 + 256 + n];
      float gg = gates[b*G4 + 512 + n];
      float go = gates[b*G4 + 768 + n];
      float cc = sigmoidf_(gf)*cst[i] + sigmoidf_(gi)*tanhf(gg);
      cst[i] = cc;
      float hh = sigmoidf_(go)*tanhf(cc);
      hbuf[b*256 + n] = f2bf(hh);
      features[((long long)(b*S_ + t))*(2*HID_) + dir*HID_ + n] = hh;
    }
    __syncthreads();
  }
}

// ---------- Elementwise / reduction kernels ----------
__global__ void cvt_bf16_k(const float* __restrict__ in, u16* __restrict__ out, int n) {
  int i = blockIdx.x * 256 + threadIdx.x;
  if (i < n) out[i] = f2bf(in[i]);
}
__global__ void bias_sum_k(const float* __restrict__ a, const float* __restrict__ b,
                           float* __restrict__ o, int n) {
  int i = blockIdx.x * 256 + threadIdx.x;
  if (i < n) o[i] = a[i] + b[i];
}
__global__ void broadcast_k(const float* __restrict__ src, float* __restrict__ dst,
                            int rep, int n) {
  int i = blockIdx.x * 256 + threadIdx.x;
  if (i < rep * n) dst[i] = src[i % n];
}
__global__ __launch_bounds__(256) void softmax_rows(float* __restrict__ X, int ncols) {
  __shared__ float red[256];
  float* p = X + (long long)blockIdx.x * ncols;
  float m = -3.0e38f;
  for (int c = threadIdx.x; c < ncols; c += 256) m = fmaxf(m, p[c]);
  red[threadIdx.x] = m; __syncthreads();
  for (int s = 128; s > 0; s >>= 1) { if (threadIdx.x < s) red[threadIdx.x] = fmaxf(red[threadIdx.x], red[threadIdx.x + s]); __syncthreads(); }
  m = red[0]; __syncthreads();
  float sum = 0.f;
  for (int c = threadIdx.x; c < ncols; c += 256) { float e = __expf(p[c] - m); p[c] = e; sum += e; }
  red[threadIdx.x] = sum; __syncthreads();
  for (int s = 128; s > 0; s >>= 1) { if (threadIdx.x < s) red[threadIdx.x] += red[threadIdx.x + s]; __syncthreads(); }
  float inv = 1.0f / red[0];
  for (int c = threadIdx.x; c < ncols; c += 256) p[c] *= inv;
}
__global__ __launch_bounds__(256) void layernorm_rows(
    const float* __restrict__ X, const float* __restrict__ g,
    const float* __restrict__ b, float* __restrict__ Y) {
  __shared__ float red[256];
  const float* p = X + (long long)blockIdx.x * D_;
  float* q = Y + (long long)blockIdx.x * D_;
  float s = 0.f;
  for (int c = threadIdx.x; c < D_; c += 256) s += p[c];
  red[threadIdx.x] = s; __syncthreads();
  for (int st = 128; st > 0; st >>= 1) { if (threadIdx.x < st) red[threadIdx.x] += red[threadIdx.x + st]; __syncthreads(); }
  float mu = red[0] / D_; __syncthreads();
  float v = 0.f;
  for (int c = threadIdx.x; c < D_; c += 256) { float d = p[c] - mu; v += d * d; }
  red[threadIdx.x] = v; __syncthreads();
  for (int st = 128; st > 0; st >>= 1) { if (threadIdx.x < st) red[threadIdx.x] += red[threadIdx.x + st]; __syncthreads(); }
  float rstd = rsqrtf(red[0] / D_ + 1e-5f);
  for (int c = threadIdx.x; c < D_; c += 256) q[c] = (p[c] - mu) * rstd * g[c] + b[c];
}
__global__ __launch_bounds__(32) void out_proj_k(
    const float* __restrict__ q, const float* __restrict__ W, float* __restrict__ out) {
  int bh = blockIdx.x, lane = threadIdx.x;
  const float* row = q + (long long)bh * D_;
  int h = bh & (H_ - 1);
  float s = 0.f;
  for (int d = lane; d < D_; d += 32) s += row[d] * W[d * H_ + h];
#pragma unroll
  for (int off = 16; off > 0; off >>= 1) s += __shfl_down(s, off, 32);
  if (lane == 0) out[bh] = s;
}

// ---------- Orchestration ----------
extern "C" void kernel_launch(void* const* d_in, const int* in_sizes, int n_in,
                              void* d_out, int out_size, void* d_ws, size_t ws_size,
                              hipStream_t stream) {
  const float* x     = (const float*)d_in[0];
  const float* lab   = (const float*)d_in[1];
  const float* oproj = (const float*)d_in[2];
  const float* Wih_f = (const float*)d_in[3];
  const float* Whh_f = (const float*)d_in[4];
  const float* bih_f = (const float*)d_in[5];
  const float* bhh_f = (const float*)d_in[6];
  const float* Wih_b = (const float*)d_in[7];
  const float* Whh_b = (const float*)d_in[8];
  const float* bih_b = (const float*)d_in[9];
  const float* bhh_b = (const float*)d_in[10];
  const float* fw1   = (const float*)d_in[11];
  const float* fb1   = (const float*)d_in[12];
  const float* fw2   = (const float*)d_in[13];
  const float* fb2   = (const float*)d_in[14];
  const float* ln_g  = (const float*)d_in[15];
  const float* ln_b  = (const float*)d_in[16];
  float* out = (float*)d_out;

  char* p = (char*)d_ws;
  auto carve = [&](size_t bytes) { char* r = p; p += (bytes + 255) & ~size_t(255); return r; };
  float* pre_f  = (float*)carve(4096ull * 1024 * 4);
  float* pre_b  = (float*)carve(4096ull * 1024 * 4);
  float* bsum_f = (float*)carve(1024 * 4);
  float* bsum_b = (float*)carve(1024 * 4);
  u16*   whhf16 = (u16*)  carve(1024ull * 256 * 2);
  u16*   whhb16 = (u16*)  carve(1024ull * 256 * 2);
  float* feat   = (float*)carve(16ull * 256 * 512 * 4);
  float* query  = (float*)carve(16ull * 1024 * 512 * 4);
  float* scores = (float*)carve(16ull * 1024 * 1024 * 4);
  float* q1     = (float*)carve(16ull * 1024 * 512 * 4);
  float* s2     = (float*)carve(16ull * 1024 * 256 * 4);
  float* q2     = (float*)carve(16ull * 1024 * 512 * 4);
  float* h1     = (float*)carve(16384ull * 2048 * 4);
  float* h2     = (float*)carve(16384ull * 512 * 4);

  bias_sum_k<<<4, 256, 0, stream>>>(bih_f, bhh_f, bsum_f, 1024);
  bias_sum_k<<<4, 256, 0, stream>>>(bih_b, bhh_b, bsum_b, 1024);
  cvt_bf16_k<<<1024, 256, 0, stream>>>(Whh_f, whhf16, 1024 * 256);
  cvt_bf16_k<<<1024, 256, 0, stream>>>(Whh_b, whhb16, 1024 * 256);

  // pre = X(4096x512) @ Wih^T(512x1024) + (bih+bhh)
  gemm_wmma<true, 1><<<dim3(32, 8, 1), 256, 0, stream>>>(x, Wih_f, bsum_f, pre_f, 4096, 1024, 512, 0, 0, 0);
  gemm_wmma<true, 1><<<dim3(32, 8, 1), 256, 0, stream>>>(x, Wih_b, bsum_b, pre_b, 4096, 1024, 512, 0, 0, 0);

  // bidirectional LSTM scan -> features (B,S,512)
  lstm_scan<<<dim3(2), dim3(1024), 16 * 1024 * 4 + 16 * 256 * 2, stream>>>(pre_f, pre_b, whhf16, whhb16, feat);

  // query = broadcast(label_embeddings) to (B,H,D)
  broadcast_k<<<32768, 256, 0, stream>>>(lab, query, B_, H_ * D_);

  for (int l = 0; l < L_; ++l) {
    // scores = Q @ Q^T  (batched 1024x1024x512)
    gemm_wmma<true, 0><<<dim3(8, 8, 16), 256, 0, stream>>>(query, query, nullptr, scores, 1024, 1024, 512, 524288, 524288, 1048576);
    softmax_rows<<<16384, 256, 0, stream>>>(scores, 1024);
    // q1 = W @ Q  (1024x512x1024)
    gemm_wmma<false, 0><<<dim3(8, 4, 16), 256, 0, stream>>>(scores, query, nullptr, q1, 1024, 512, 1024, 1048576, 524288, 524288);
    // s2 = q1 @ feat^T  (1024x256x512)
    gemm_wmma<true, 0><<<dim3(8, 2, 16), 256, 0, stream>>>(q1, feat, nullptr, s2, 1024, 256, 512, 524288, 131072, 262144);
    softmax_rows<<<16384, 256, 0, stream>>>(s2, 256);
    // q2 = w2 @ feat  (1024x512x256)
    gemm_wmma<false, 0><<<dim3(8, 4, 16), 256, 0, stream>>>(s2, feat, nullptr, q2, 1024, 512, 256, 262144, 131072, 524288);
    // FFN: h1 = gelu(q2 @ w1 + b1)  (16384x2048x512), h2 = h1 @ w2 + b2 (16384x512x2048)
    gemm_wmma<false, 2><<<dim3(128, 16, 1), 256, 0, stream>>>(q2, fw1 + (size_t)l * 512 * 2048, fb1 + l * 2048, h1, 16384, 2048, 512, 0, 0, 0);
    gemm_wmma<false, 1><<<dim3(128, 4, 1), 256, 0, stream>>>(h1, fw2 + (size_t)l * 2048 * 512, fb2 + l * 512, h2, 16384, 512, 2048, 0, 0, 0);
    layernorm_rows<<<16384, 256, 0, stream>>>(h2, ln_g + l * 512, ln_b + l * 512, query);
  }
  out_proj_k<<<16384, 32, 0, stream>>>(query, oproj, out);
}
